// DecoderSingleCrossAttn_82145544503488
// MI455X (gfx1250) — compile-verified, matrix operand order checked by reading, not run on hardware
//
#include <hip/hip_runtime.h>
#include <math.h>

// ---------------------------------------------------------------------------
// MI455X (gfx1250, wave32) Mamba-decoder-with-cross-attention pipeline.
// Dense GEMMs: v_wmma_f32_16x16x32_f16 (f16 in, f32 accumulate), double-
// buffered LDS staging with vectorized b128 global loads.
// Elementwise stages (conv, scan, norms, softmax) are VALU kernels.
// ---------------------------------------------------------------------------

typedef __attribute__((ext_vector_type(16))) _Float16 v16h;
typedef __attribute__((ext_vector_type(2)))  __fp16   v2fp;   // cvt_pkrtz return type
typedef __attribute__((ext_vector_type(8)))  float    v8f;
typedef __attribute__((ext_vector_type(4)))  float    v4f;

#define BATCH   8
#define SEQ_T   1024
#define SEQ_S   256
#define ROWS    (BATCH * SEQ_T)   /* 8192 */
#define KV_ROWS (BATCH * SEQ_S)   /* 2048 */
#define DMODEL  512
#define DINNER  1024
#define DSTATE  16
#define DTRANK  32
#define NMELS   80

enum { EP_NONE = 0, EP_BIAS = 1, EP_BIAS_SILU = 2, EP_BIAS_SOFTPLUS = 3,
       EP_RESID = 4, EP_BIAS_RESID = 5 };

struct GemmP {
    const float* A;
    const float* W;
    float*       C;
    const float* bias;
    const float* resid;
    int M, N, K, lda, ldb, ldc;
    long long sA1, sB1, sC1;   // batch-dim-1 strides
    long long sA2, sB2, sC2;   // batch-dim-2 strides
    int nb2;                   // inner batch count (grid.z = nb1*nb2)
    int wTrans;                // 0: W[k*ldb+n]   1: W[n*ldb+k]  (i.e. A @ W.T)
    int ep;                    // epilogue selector
};

__device__ __forceinline__ unsigned int pack2(float a, float b) {
    union { v2fp h; unsigned int u; } x;
    x.h = __builtin_amdgcn_cvt_pkrtz(a, b);   // v_cvt_pk_rtz_f16_f32
    return x.u;
}

// ---------------------------------------------------------------------------
// Tiled WMMA GEMM: block = 256 threads = 8 waves, tile 128(M) x 64(N),
// K stepped by 32, double-buffered LDS (one barrier per k-step).
// Requires M % 128 == 0 (true for every call in this pipeline).
// Fragments assembled per CDNA5 ISA 16-bit A/B VGPR layouts.
// ---------------------------------------------------------------------------
__global__ __launch_bounds__(256) void gemm_wmma(GemmP p) {
    __shared__ alignas(16) _Float16 As[2][128 * 34];
    __shared__ alignas(16) _Float16 Bs[2][64 * 34];

    const int tid  = threadIdx.x;
    const int lane = tid & 31;
    const int wv   = tid >> 5;
    const int wr   = wv >> 1;          // wave row (0..3) -> 32 rows each
    const int wc   = wv & 1;           // wave col (0..1) -> 32 cols each
    const int hl   = lane >> 4;        // lane half-select
    const int l16  = lane & 15;

    const int z  = blockIdx.z;
    const int z1 = z / p.nb2;
    const int z2 = z % p.nb2;
    const float* A = p.A + (long long)z1 * p.sA1 + (long long)z2 * p.sA2;
    const float* W = p.W + (long long)z1 * p.sB1 + (long long)z2 * p.sB2;
    float*       C = p.C + (long long)z1 * p.sC1 + (long long)z2 * p.sC2;
    const float* R = p.resid ? (p.resid + (long long)z1 * p.sC1 + (long long)z2 * p.sC2)
                             : (const float*)0;

    const int tm = blockIdx.y * 128;
    const int tn = blockIdx.x * 64;

    v8f acc[2][2];
    for (int i = 0; i < 2; i++)
        for (int j = 0; j < 2; j++)
            acc[i][j] = (v8f){0.f, 0.f, 0.f, 0.f, 0.f, 0.f, 0.f, 0.f};

    union U16 { v16h v; unsigned int u[8]; };

    // Per-thread staging registers for one k-slab
    v4f aR[4];   // A: 128x32 = 4096 f32 -> 4 x float4 per thread
    v4f bR[2];   // B:  64x32 = 2048 f32 -> 2 x float4 per thread

    // ---- load one k-slab (global -> regs), fast path = unguarded b128 ----
    auto load_tiles = [&](int kk) {
        const bool fk = (kk + 32 <= p.K);
        // A tile: thread j-th chunk -> row m = i>>3, quad kq = (i&7)*4
        for (int j = 0; j < 4; j++) {
            int i = tid + j * 256;
            int m = i >> 3, kq = (i & 7) << 2;
            long long base = (long long)(tm + m) * p.lda + kk + kq;
            if (fk) {
                aR[j] = *(const v4f*)&A[base];
            } else {
                v4f t = {0.f, 0.f, 0.f, 0.f};
                for (int c = 0; c < 4; c++)
                    if (kk + kq + c < p.K) t[c] = A[base + c];
                aR[j] = t;
            }
        }
        const bool fb = fk && (tn + 64 <= p.N);
        if (p.wTrans) {
            // rows are n, contiguous along k
            for (int j = 0; j < 2; j++) {
                int i = tid + j * 256;
                int n = i >> 3, kq = (i & 7) << 2;
                int gn = tn + n;
                long long base = (long long)gn * p.ldb + kk + kq;
                if (fb) {
                    bR[j] = *(const v4f*)&W[base];
                } else {
                    v4f t = {0.f, 0.f, 0.f, 0.f};
                    if (gn < p.N)
                        for (int c = 0; c < 4; c++)
                            if (kk + kq + c < p.K) t[c] = W[base + c];
                    bR[j] = t;
                }
            }
        } else {
            // rows are k, contiguous along n
            for (int j = 0; j < 2; j++) {
                int i = tid + j * 256;
                int k = i >> 4, nq = (i & 15) << 2;
                int gk = kk + k;
                long long base = (long long)gk * p.ldb + tn + nq;
                if (fb) {
                    bR[j] = *(const v4f*)&W[base];
                } else {
                    v4f t = {0.f, 0.f, 0.f, 0.f};
                    if (gk < p.K)
                        for (int c = 0; c < 4; c++)
                            if (tn + nq + c < p.N) t[c] = W[base + c];
                    bR[j] = t;
                }
            }
        }
    };

    // ---- regs -> LDS (f16, padded rows; dword stores) ----
    auto store_tiles = [&](int buf) {
        for (int j = 0; j < 4; j++) {
            int i = tid + j * 256;
            int m = i >> 3, kq = (i & 7) << 2;
            unsigned int* d = (unsigned int*)&As[buf][m * 34 + kq];
            d[0] = pack2(aR[j][0], aR[j][1]);
            d[1] = pack2(aR[j][2], aR[j][3]);
        }
        if (p.wTrans) {
            for (int j = 0; j < 2; j++) {
                int i = tid + j * 256;
                int n = i >> 3, kq = (i & 7) << 2;
                unsigned int* d = (unsigned int*)&Bs[buf][n * 34 + kq];
                d[0] = pack2(bR[j][0], bR[j][1]);
                d[1] = pack2(bR[j][2], bR[j][3]);
            }
        } else {
            // transpose on store: 4 consecutive n, same k
            for (int j = 0; j < 2; j++) {
                int i = tid + j * 256;
                int k = i >> 4, nq = (i & 15) << 2;
                for (int c = 0; c < 4; c++)
                    Bs[buf][(nq + c) * 34 + k] = (_Float16)bR[j][c];
            }
        }
    };

    const int nk = (p.K + 31) >> 5;
    load_tiles(0);
    store_tiles(0);
    int buf = 0;
    for (int it = 0; it < nk; it++) {
        __syncthreads();                       // LDS[buf] ready for all
        if (it + 1 < nk) load_tiles((it + 1) << 5);   // overlap with compute

        // ---- fragments per ISA 16-bit layouts: k0 = (r/4)*16 + hl*8 + (r%4)*2
        U16 af[2], bf[2];
        for (int t = 0; t < 2; t++) {
            for (int r = 0; r < 8; r++) {
                int k0 = ((r >> 2) << 4) + (hl << 3) + ((r & 3) << 1);
                af[t].u[r] = *(const unsigned int*)&As[buf][(wr * 32 + t * 16 + l16) * 34 + k0];
                bf[t].u[r] = *(const unsigned int*)&Bs[buf][(wc * 32 + t * 16 + l16) * 34 + k0];
            }
        }
        for (int mi = 0; mi < 2; mi++)
            for (int ni = 0; ni < 2; ni++)
                acc[mi][ni] = __builtin_amdgcn_wmma_f32_16x16x32_f16(
                    false, af[mi].v, false, bf[ni].v,
                    (short)0, acc[mi][ni], false, false);

        if (it + 1 < nk) store_tiles(buf ^ 1); // safe: buf^1 was last read in the previous k-step, before this step's barrier
        buf ^= 1;
    }

    // ---- epilogue: C/D layout -> m = r + 8*half, n = lane&15 ----
    for (int mi = 0; mi < 2; mi++)
        for (int ni = 0; ni < 2; ni++)
            for (int r = 0; r < 8; r++) {
                int gm = tm + wr * 32 + mi * 16 + r + 8 * hl;
                int gn = tn + wc * 32 + ni * 16 + l16;
                if (gm < p.M && gn < p.N) {
                    float fv = acc[mi][ni][r];
                    if (p.ep == EP_BIAS || p.ep == EP_BIAS_SILU ||
                        p.ep == EP_BIAS_SOFTPLUS || p.ep == EP_BIAS_RESID)
                        fv += p.bias[gn];
                    if (p.ep == EP_BIAS_SILU)
                        fv = fv / (1.f + __expf(-fv));
                    if (p.ep == EP_BIAS_SOFTPLUS)
                        fv = (fv > 20.f) ? fv : log1pf(__expf(fv));
                    if (p.ep == EP_RESID || p.ep == EP_BIAS_RESID)
                        fv += R[(long long)gm * p.ldc + gn];
                    C[(long long)gm * p.ldc + gn] = fv;
                }
            }
}

// ---------------------------------------------------------------------------
// Row normalizations: one 256-thread block per row (D = 512).
// ---------------------------------------------------------------------------
__global__ __launch_bounds__(256) void rmsnorm_kernel(const float* __restrict__ x,
                                                      const float* __restrict__ w,
                                                      float* __restrict__ out, int D) {
    __shared__ float red[256];
    long long row = blockIdx.x;
    const float* xr = x + row * D;
    float s = 0.f;
    for (int i = threadIdx.x; i < D; i += 256) { float v = xr[i]; s += v * v; }
    red[threadIdx.x] = s; __syncthreads();
    for (int off = 128; off > 0; off >>= 1) {
        if (threadIdx.x < off) red[threadIdx.x] += red[threadIdx.x + off];
        __syncthreads();
    }
    float inv = rsqrtf(red[0] / (float)D + 1e-5f);
    for (int i = threadIdx.x; i < D; i += 256)
        out[row * D + i] = xr[i] * inv * w[i];
}

__global__ __launch_bounds__(256) void layernorm_kernel(const float* __restrict__ x,
                                                        const float* __restrict__ w,
                                                        const float* __restrict__ b,
                                                        float* __restrict__ out, int D) {
    __shared__ float rs[256], rq[256];
    long long row = blockIdx.x;
    const float* xr = x + row * D;
    float s = 0.f, q = 0.f;
    for (int i = threadIdx.x; i < D; i += 256) { float v = xr[i]; s += v; q += v * v; }
    rs[threadIdx.x] = s; rq[threadIdx.x] = q; __syncthreads();
    for (int off = 128; off > 0; off >>= 1) {
        if (threadIdx.x < off) { rs[threadIdx.x] += rs[threadIdx.x + off];
                                 rq[threadIdx.x] += rq[threadIdx.x + off]; }
        __syncthreads();
    }
    float mu  = rs[0] / (float)D;
    float var = rq[0] / (float)D - mu * mu;
    float inv = rsqrtf(var + 1e-5f);
    for (int i = threadIdx.x; i < D; i += 256)
        out[row * D + i] = (xr[i] - mu) * inv * w[i] + b[i];
}

// ---------------------------------------------------------------------------
// Softmax over 256-wide rows with 1/sqrt(64) scale (attention scores).
// ---------------------------------------------------------------------------
__global__ __launch_bounds__(256) void softmax256(float* __restrict__ s) {
    __shared__ float red[256];
    long long row = blockIdx.x;
    float* r = s + row * 256;
    float v = r[threadIdx.x] * 0.125f;
    red[threadIdx.x] = v; __syncthreads();
    for (int off = 128; off > 0; off >>= 1) {
        if (threadIdx.x < off)
            red[threadIdx.x] = fmaxf(red[threadIdx.x], red[threadIdx.x + off]);
        __syncthreads();
    }
    float mx = red[0]; __syncthreads();
    float e = __expf(v - mx);
    red[threadIdx.x] = e; __syncthreads();
    for (int off = 128; off > 0; off >>= 1) {
        if (threadIdx.x < off) red[threadIdx.x] += red[threadIdx.x + off];
        __syncthreads();
    }
    r[threadIdx.x] = e / red[0];
}

// ---------------------------------------------------------------------------
// Depthwise causal conv1d (width 4) + SiLU. xz holds [xi | z] (ld 2048).
// ---------------------------------------------------------------------------
__global__ __launch_bounds__(256) void conv_silu(const float* __restrict__ xz,
                                                 const float* __restrict__ cw,
                                                 const float* __restrict__ cb,
                                                 float* __restrict__ xc) {
    long long idx = (long long)blockIdx.x * 256 + threadIdx.x;
    if (idx >= (long long)ROWS * DINNER) return;
    int d       = (int)(idx & (DINNER - 1));
    long long r = idx >> 10;                 // global row = b*T + t
    int t       = (int)(r & (SEQ_T - 1));
    float acc = cb[d];
    for (int k = 0; k < 4; k++) {
        int tt = t - 3 + k;
        if (tt >= 0)
            acc += xz[(r - (3 - k)) * (2 * DINNER) + d] * cw[d * 4 + k];
    }
    xc[idx] = acc / (1.f + __expf(-acc));
}

// ---------------------------------------------------------------------------
// Selective scan: thread = one (b, d) channel; h[16], A[16] in registers;
// per-step B/C vectors staged through LDS; fused +u*D and *silu(z) gating.
// ---------------------------------------------------------------------------
__global__ __launch_bounds__(256) void scan_kernel(const float* __restrict__ xc,
                                                   const float* __restrict__ delta,
                                                   const float* __restrict__ xdbl,
                                                   const float* __restrict__ xz,
                                                   const float* __restrict__ A_log,
                                                   const float* __restrict__ Dv,
                                                   float* __restrict__ g) {
    __shared__ float sB[DSTATE], sC[DSTATE];
    const int d = blockIdx.x * 256 + threadIdx.x;
    const int b = blockIdx.y;
    float Ar[DSTATE], h[DSTATE];
    for (int n = 0; n < DSTATE; n++) {
        Ar[n] = -__expf(A_log[d * DSTATE + n]);
        h[n]  = 0.f;
    }
    const float Dd = Dv[d];
    for (int t = 0; t < SEQ_T; t++) {
        long long row = (long long)b * SEQ_T + t;
        if (threadIdx.x < DSTATE)
            sB[threadIdx.x] = xdbl[row * 64 + DTRANK + threadIdx.x];
        else if (threadIdx.x < 2 * DSTATE)
            sC[threadIdx.x - DSTATE] = xdbl[row * 64 + DTRANK + DSTATE + (threadIdx.x - DSTATE)];
        __syncthreads();
        float u  = xc[row * DINNER + d];
        float dl = delta[row * DINNER + d];
        float zz = xz[row * (2 * DINNER) + DINNER + d];
        float y = 0.f;
#pragma unroll
        for (int n = 0; n < DSTATE; n++) {
            float dA = __expf(dl * Ar[n]);
            h[n] = dA * h[n] + (dl * sB[n]) * u;
            y += h[n] * sC[n];
        }
        float sz = zz / (1.f + __expf(-zz));
        g[row * DINNER + d] = (y + u * Dd) * sz;
        __syncthreads();
    }
}

// ---------------------------------------------------------------------------
// Host-side helpers
// ---------------------------------------------------------------------------
static void gemm(hipStream_t st, const float* A, const float* W, float* C,
                 const float* bias, const float* resid,
                 int M, int N, int K, int lda, int ldb, int ldc,
                 int wTrans, int ep,
                 int nb1 = 1, int nb2 = 1,
                 long long sA1 = 0, long long sB1 = 0, long long sC1 = 0,
                 long long sA2 = 0, long long sB2 = 0, long long sC2 = 0) {
    GemmP p;
    p.A = A; p.W = W; p.C = C; p.bias = bias; p.resid = resid;
    p.M = M; p.N = N; p.K = K; p.lda = lda; p.ldb = ldb; p.ldc = ldc;
    p.sA1 = sA1; p.sB1 = sB1; p.sC1 = sC1;
    p.sA2 = sA2; p.sB2 = sB2; p.sC2 = sC2;
    p.nb2 = nb2; p.wTrans = wTrans; p.ep = ep;
    dim3 grid((N + 63) / 64, (M + 127) / 128, nb1 * nb2);
    gemm_wmma<<<grid, dim3(256), 0, st>>>(p);
}

static void run_mamba(hipStream_t st, float* h, float* hB, float* xz, float* xc,
                      float* xdbl, float* delta, float* g,
                      const float* norm_w, const float* in_w,
                      const float* cw, const float* cb,
                      const float* xw, const float* dtw, const float* dtb,
                      const float* alog, const float* Dv, const float* ow) {
    rmsnorm_kernel<<<ROWS, 256, 0, st>>>(h, norm_w, hB, DMODEL);
    gemm(st, hB, in_w, xz, 0, 0, ROWS, 2 * DINNER, DMODEL, DMODEL, 2 * DINNER, 2 * DINNER, 0, EP_NONE);
    conv_silu<<<(ROWS * DINNER) / 256, 256, 0, st>>>(xz, cw, cb, xc);
    gemm(st, xc, xw, xdbl, 0, 0, ROWS, 64, DINNER, DINNER, 64, 64, 0, EP_NONE);
    gemm(st, xdbl, dtw, delta, dtb, 0, ROWS, DINNER, DTRANK, 64, DINNER, DINNER, 0, EP_BIAS_SOFTPLUS);
    scan_kernel<<<dim3(DINNER / 256, BATCH), 256, 0, st>>>(xc, delta, xdbl, xz, alog, Dv, g);
    gemm(st, g, ow, h, 0, h, ROWS, DMODEL, DINNER, DINNER, DMODEL, DMODEL, 0, EP_RESID);
}

extern "C" void kernel_launch(void* const* d_in, const int* in_sizes, int n_in,
                              void* d_out, int out_size, void* d_ws, size_t ws_size,
                              hipStream_t stream) {
    (void)in_sizes; (void)n_in; (void)out_size; (void)ws_size;

    const float* x             = (const float*)d_in[0];
    const float* enc           = (const float*)d_in[1];
    const float* input_proj_w  = (const float*)d_in[2];
    const float* input_proj_b  = (const float*)d_in[3];
    const float* attn_in_w     = (const float*)d_in[4];
    const float* attn_in_b     = (const float*)d_in[5];
    const float* attn_out_w    = (const float*)d_in[6];
    const float* attn_out_b    = (const float*)d_in[7];
    const float* ln_w          = (const float*)d_in[8];
    const float* ln_b          = (const float*)d_in[9];
    const float* norm_f_w      = (const float*)d_in[10];
    const float* output_w      = (const float*)d_in[11];
    const float* blk_in_proj_w = (const float*)d_in[12];
    const float* blk_conv_w    = (const float*)d_in[13];
    const float* blk_conv_b    = (const float*)d_in[14];
    const float* blk_x_proj_w  = (const float*)d_in[15];
    const float* blk_dt_proj_w = (const float*)d_in[16];
    const float* blk_dt_proj_b = (const float*)d_in[17];
    const float* blk_A_log     = (const float*)d_in[18];
    const float* blk_D         = (const float*)d_in[19];
    const float* blk_out_proj_w= (const float*)d_in[20];
    const float* blk_norm_w    = (const float*)d_in[21];
    float* out = (float*)d_out;

    // --- workspace layout (floats) ---
    float* ws    = (float*)d_ws;
    float* hA    = ws;                               // 8192*512
    float* hB    = hA   + (size_t)ROWS * DMODEL;     // 8192*512
    float* xz    = hB   + (size_t)ROWS * DMODEL;     // 8192*2048
    float* xc    = xz   + (size_t)ROWS * 2 * DINNER; // 8192*1024
    float* xdbl  = xc   + (size_t)ROWS * DINNER;     // 8192*64
    float* delta = xdbl + (size_t)ROWS * 64;         // 8192*1024
    float* g     = delta+ (size_t)ROWS * DINNER;     // 8192*1024
    // attention-phase aliases (xz / xc / g are free then)
    float* Q  = g;                                   // 8192*512
    float* Km = g + (size_t)ROWS * DMODEL;           // 2048*512
    float* Vm = Km + (size_t)KV_ROWS * DMODEL;       // 2048*512
    float* sc = xz;                                  // 64*1024*256 == 8192*2048
    float* o  = xc;                                  // 8192*512

    // 1) input projection: hA = x @ input_proj_w + b
    gemm(stream, x, input_proj_w, hA, input_proj_b, 0,
         ROWS, DMODEL, NMELS, NMELS, DMODEL, DMODEL, 0, EP_BIAS);

    // 2) first 4 Mamba blocks
    for (int i = 0; i < 4; i++)
        run_mamba(stream, hA, hB, xz, xc, xdbl, delta, g,
                  blk_norm_w + i * DMODEL,
                  blk_in_proj_w + (size_t)i * DMODEL * 2 * DINNER,
                  blk_conv_w + (size_t)i * DINNER * 4,
                  blk_conv_b + (size_t)i * DINNER,
                  blk_x_proj_w + (size_t)i * DINNER * 64,
                  blk_dt_proj_w + (size_t)i * DTRANK * DINNER,
                  blk_dt_proj_b + (size_t)i * DINNER,
                  blk_A_log + (size_t)i * DINNER * DSTATE,
                  blk_D + (size_t)i * DINNER,
                  blk_out_proj_w + (size_t)i * DINNER * DMODEL);

    // 3) cross attention (8 heads, hd=64; weights are applied as @ W.T)
    gemm(stream, hA,  attn_in_w,                     Q,  attn_in_b,        0,
         ROWS,    DMODEL, DMODEL, DMODEL, DMODEL, DMODEL, 1, EP_BIAS);
    gemm(stream, enc, attn_in_w + 1 * DMODEL*DMODEL, Km, attn_in_b + 512,  0,
         KV_ROWS, DMODEL, DMODEL, DMODEL, DMODEL, DMODEL, 1, EP_BIAS);
    gemm(stream, enc, attn_in_w + 2 * DMODEL*DMODEL, Vm, attn_in_b + 1024, 0,
         KV_ROWS, DMODEL, DMODEL, DMODEL, DMODEL, DMODEL, 1, EP_BIAS);
    // scores[b,h] = Q_slice(1024x64) @ K_slice^T(64x256)
    gemm(stream, Q, Km, sc, 0, 0,
         SEQ_T, SEQ_S, 64, DMODEL, DMODEL, SEQ_S, 1, EP_NONE,
         BATCH, 8,
         (long long)SEQ_T * DMODEL, (long long)SEQ_S * DMODEL, (long long)8 * SEQ_T * SEQ_S,
         64, 64, (long long)SEQ_T * SEQ_S);
    softmax256<<<BATCH * 8 * SEQ_T, 256, 0, stream>>>(sc);
    // o[b,h] = attn(1024x256) @ V_slice(256x64)
    gemm(stream, sc, Vm, o, 0, 0,
         SEQ_T, 64, SEQ_S, SEQ_S, DMODEL, DMODEL, 0, EP_NONE,
         BATCH, 8,
         (long long)8 * SEQ_T * SEQ_S, (long long)SEQ_S * DMODEL, (long long)SEQ_T * DMODEL,
         (long long)SEQ_T * SEQ_S, 64, 64);
    // hB = o @ attn_out_w.T + b + hA ; hA = layernorm(hB)
    gemm(stream, o, attn_out_w, hB, attn_out_b, hA,
         ROWS, DMODEL, DMODEL, DMODEL, DMODEL, DMODEL, 1, EP_BIAS_RESID);
    layernorm_kernel<<<ROWS, 256, 0, stream>>>(hB, ln_w, ln_b, hA, DMODEL);

    // 4) last 4 Mamba blocks
    for (int i = 4; i < 8; i++)
        run_mamba(stream, hA, hB, xz, xc, xdbl, delta, g,
                  blk_norm_w + i * DMODEL,
                  blk_in_proj_w + (size_t)i * DMODEL * 2 * DINNER,
                  blk_conv_w + (size_t)i * DINNER * 4,
                  blk_conv_b + (size_t)i * DINNER,
                  blk_x_proj_w + (size_t)i * DINNER * 64,
                  blk_dt_proj_w + (size_t)i * DTRANK * DINNER,
                  blk_dt_proj_b + (size_t)i * DINNER,
                  blk_A_log + (size_t)i * DINNER * DSTATE,
                  blk_D + (size_t)i * DINNER,
                  blk_out_proj_w + (size_t)i * DINNER * DMODEL);

    // 5) final norm + output projection
    rmsnorm_kernel<<<ROWS, 256, 0, stream>>>(hA, norm_f_w, hB, DMODEL);
    gemm(stream, hB, output_w, out, 0, 0,
         ROWS, NMELS, DMODEL, DMODEL, NMELS, NMELS, 0, EP_NONE);
}
